// BertSelfAttention_42356967473599
// MI455X (gfx1250) — matrix operand order repaired and checked by reading, not compiled, not run on hardware
//
#include <hip/hip_runtime.h>
#include <hip/hip_bf16.h>

// ---------------------------------------------------------------------------
// BERT self-attention for MI455X (gfx1250, wave32, WMMA bf16 16x16x32 + TDM).
// B=2, S=2048, H=1024, nh=16, hd=64.
// ---------------------------------------------------------------------------

typedef __bf16 bf16_t;
typedef __attribute__((ext_vector_type(16))) __bf16 v16bf;
typedef __attribute__((ext_vector_type(8)))  float  v8f;
typedef __attribute__((ext_vector_type(4)))  unsigned int u32x4;
typedef __attribute__((ext_vector_type(8)))  int          i32x8;
typedef __attribute__((ext_vector_type(4)))  int          i32x4;

#define USE_TDM 1

union FragAB {           // 16 bf16 = 8 dwords (A 16x32 or B 32x16 fragment)
    v16bf v;
    unsigned int u[8];
};
union FragC {            // 8 f32 accumulator fragment (C/D 16x16)
    v8f v;
    float f[8];
};

__device__ __forceinline__ v8f wmma_bf16(const FragAB& a, const FragAB& b, const v8f& c) {
    return __builtin_amdgcn_wmma_f32_16x16x32_bf16(false, a.v, false, b.v,
                                                   (short)0, c, false, false);
}

// A-fragment K-offset (contiguous order, matching B fills):
// lanes 0-15 hold K pairs {0,2,...,14}; lanes 16-31 hold +16.
__device__ __forceinline__ int a_koff(int j, int hi) {
    return hi * 16 + 2 * j;
}

// ---------------------------------------------------------------------------
// Kernel 1: fp32 -> bf16 conversion (grid-stride)
// ---------------------------------------------------------------------------
__global__ void cvt_f32_bf16(const float* __restrict__ src, bf16_t* __restrict__ dst, int n) {
    for (int i = blockIdx.x * blockDim.x + threadIdx.x; i < n; i += gridDim.x * blockDim.x)
        dst[i] = (bf16_t)src[i];
}

// ---------------------------------------------------------------------------
// Kernel 2: fused QKV projection.  out = X @ W^T + b, written head-major bf16:
//   dst[((b*16 + h)*2048 + s)*64 + d]
// grid = (BS/128, H/128, 3 {Q,K,V}), block = 256 (8 waves, 4x2 wave grid,
// wave tile 32x64 -> 8 WMMA per K-step).  Global->reg->LDS software pipeline.
// ---------------------------------------------------------------------------
#define GLDX 34   // padded LDS row (bf16 elems): 68B = 17 dwords -> conflict-free
__global__ __launch_bounds__(256) void qkv_gemm(
    const bf16_t* __restrict__ Xb,
    const bf16_t* __restrict__ Wqb, const bf16_t* __restrict__ Wkb, const bf16_t* __restrict__ Wvb,
    const float* __restrict__ bq, const float* __restrict__ bk, const float* __restrict__ bv,
    bf16_t* __restrict__ Qo, bf16_t* __restrict__ Ko, bf16_t* __restrict__ Vo)
{
    __shared__ bf16_t Xs[128][GLDX];
    __shared__ bf16_t Ws[128][GLDX];

    const bf16_t* W;
    const float*  bias;
    bf16_t*       dst;
    if (blockIdx.z == 0)      { W = Wqb; bias = bq; dst = Qo; }
    else if (blockIdx.z == 1) { W = Wkb; bias = bk; dst = Ko; }
    else                      { W = Wvb; bias = bv; dst = Vo; }

    const int tid  = threadIdx.x;
    const int lane = tid & 31, wv = tid >> 5;
    const int wm = wv >> 1, wn = wv & 1;        // 4 M-subtiles(32) x 2 N-subtiles(64)
    const int lh = lane & 15, hi = lane >> 4;
    const int M0 = blockIdx.x * 128, N0 = blockIdx.y * 128;

    // staging: 256 threads x 16 bf16 per tile (two 16B global reads each)
    const int sr = tid >> 1;            // 0..127 row
    const int sc = (tid & 1) * 16;      // 0 or 16

    FragC c[2][4];
#pragma unroll
    for (int at = 0; at < 2; ++at)
#pragma unroll
        for (int nt = 0; nt < 4; ++nt)
#pragma unroll
            for (int i = 0; i < 8; ++i) c[at][nt].f[i] = 0.f;

    uint4 nx0, nx1, nw0, nw1;
    auto gload = [&](int kk) {
        const bf16_t* xg = Xb + (size_t)(M0 + sr) * 1024 + kk + sc;
        nx0 = ((const uint4*)xg)[0];
        nx1 = ((const uint4*)xg)[1];
        const bf16_t* wg = W + (size_t)(N0 + sr) * 1024 + kk + sc;
        nw0 = ((const uint4*)wg)[0];
        nw1 = ((const uint4*)wg)[1];
    };
    gload(0);

    for (int kk = 0; kk < 1024; kk += 32) {
        {   // commit prefetched slab to LDS (4B granular: padded rows)
            unsigned int* px = (unsigned int*)&Xs[sr][sc];
            px[0] = nx0.x; px[1] = nx0.y; px[2] = nx0.z; px[3] = nx0.w;
            px[4] = nx1.x; px[5] = nx1.y; px[6] = nx1.z; px[7] = nx1.w;
            unsigned int* pw = (unsigned int*)&Ws[sr][sc];
            pw[0] = nw0.x; pw[1] = nw0.y; pw[2] = nw0.z; pw[3] = nw0.w;
            pw[4] = nw1.x; pw[5] = nw1.y; pw[6] = nw1.z; pw[7] = nw1.w;
        }
        __syncthreads();
        if (kk + 32 < 1024) gload(kk + 32);   // next slab in flight during WMMA

        FragAB a[2];
#pragma unroll
        for (int at = 0; at < 2; ++at)
#pragma unroll
            for (int j = 0; j < 8; ++j)
                a[at].u[j] = *(const unsigned int*)&Xs[wm * 32 + at * 16 + lh][a_koff(j, hi)];
#pragma unroll
        for (int nt = 0; nt < 4; ++nt) {
            FragAB b;
#pragma unroll
            for (int j = 0; j < 8; ++j)
                b.u[j] = *(const unsigned int*)&Ws[wn * 64 + nt * 16 + lh][hi * 16 + 2 * j];
            c[0][nt].v = wmma_bf16(a[0], b, c[0][nt].v);
            c[1][nt].v = wmma_bf16(a[1], b, c[1][nt].v);
        }
        __syncthreads();
    }

    // epilogue: bias + head-major bf16 store
#pragma unroll
    for (int at = 0; at < 2; ++at)
#pragma unroll
    for (int nt = 0; nt < 4; ++nt) {
        const int n = N0 + wn * 64 + nt * 16 + lh;
        const int h = n >> 6, d = n & 63;
        const float bb = bias[n];
#pragma unroll
        for (int i = 0; i < 8; ++i) {
            const int m = M0 + wm * 32 + at * 16 + hi * 8 + i;  // C layout: vgpr i -> row
            const int b = m >> 11, s = m & 2047;
            dst[(((size_t)b * 16 + h) * 2048 + s) * 64 + d] = (bf16_t)(c[at][nt].f[i] + bb);
        }
    }
}

// ---------------------------------------------------------------------------
// TDM: stage one contiguous 32x64 bf16 K-tile (4KB) into LDS with padding
// (1 dword pad every 32 dwords -> padded row stride 66 bf16 = 33 dwords).
// Uses the 6-arg clang-23 builtin form:
//   (u32x4 g0, i32x8 g1, i32x4 g2, i32x4 g3, i32x8 extra, i32 cpol)
// ---------------------------------------------------------------------------
#if USE_TDM
__device__ __forceinline__ void tdm_load_ktile(const bf16_t* gsrc, unsigned int lds_byte_off) {
    const unsigned long long ga = (unsigned long long)(size_t)gsrc;
    u32x4 g0;
    g0[0] = 1u;                                   // count=1, user descriptor
    g0[1] = lds_byte_off;                         // lds_addr (bytes)
    g0[2] = (unsigned int)(ga & 0xffffffffu);     // global_addr[31:0]
    g0[3] = (unsigned int)((ga >> 32) & 0x01ffffffu) | (2u << 30);  // addr[56:32] | type=2
    i32x8 g1;
    g1[0] = (int)((1u << 16) |                    // data_size = 2B
                  (1u << 20) |                    // pad_enable
                  (4u << 22));                    // pad_interval: 32 dwords (pad_amount=0 -> 1 dword)
    g1[1] = 0;                                    // atomic addr / tensor_dim0[15:0]=0
    g1[2] = (int)((0x0100u) | (1u << 16));        // tensor_dim0=0x01000000 (hi), tensor_dim1=1 (lo)
    g1[3] = (int)(2048u << 16);                   // tile_dim0 = 2048 elements
    g1[4] = 1;                                    // tile_dim1 = 1, tile_dim2 = 0
    g1[5] = 2048;                                 // tensor_dim0_stride[31:0]
    g1[6] = 0;
    g1[7] = 0;
    const i32x4 z4 = {0, 0, 0, 0};                // groups 2/3 unused (<=2D tensor)
    const i32x8 z8 = {0, 0, 0, 0, 0, 0, 0, 0};
    __builtin_amdgcn_tensor_load_to_lds(g0, g1, z4, z4, z8, 0);
}
#endif

// ---------------------------------------------------------------------------
// Kernel 3: flash-style attention.  grid = (S/128, B*nh), block = 128 (4 waves).
// Each wave owns 32 query rows (Q frags resident); streams 32-key blocks with
// online softmax.  16 WMMA per key block per wave.
// ---------------------------------------------------------------------------
#define LDK 66   // 132B = 33 dwords -> conflict-free (and matches TDM padding)
#define LDV 34   // 68B  = 17 dwords
#define LDP 34
__global__ __launch_bounds__(128) void attn(
    const bf16_t* __restrict__ Qb, const bf16_t* __restrict__ Kb,
    const bf16_t* __restrict__ Vb, const float* __restrict__ cbias,
    float* __restrict__ out)
{
    __shared__ bf16_t Ks[32][LDK];
    __shared__ bf16_t Vt[64][LDV];        // V transposed: Vt[d][key]
    __shared__ bf16_t Ps[4][32][LDP];     // per-wave P relayout scratch

    const int tid  = threadIdx.x;
    const int lane = tid & 31, wv = tid >> 5;
    const int lh = lane & 15, hi = lane >> 4;
    const int bh = blockIdx.y;                 // b*16 + h
    const int b  = bh >> 4, h = bh & 15;
    const int q0 = blockIdx.x * 128;

    const bf16_t* Qh = Qb + (size_t)bh * 2048 * 64;
    const bf16_t* Kh = Kb + (size_t)bh * 2048 * 64;
    const bf16_t* Vh = Vb + (size_t)bh * 2048 * 64;

    // Q fragments (2 qtiles x 2 K-halves) resident for the whole pass
    FragAB qa[2][2];
#pragma unroll
    for (int qt = 0; qt < 2; ++qt) {
        const bf16_t* qrow = Qh + (size_t)(q0 + wv * 32 + qt * 16 + lh) * 64;
#pragma unroll
        for (int f = 0; f < 2; ++f)
#pragma unroll
            for (int j = 0; j < 8; ++j)
                qa[qt][f].u[j] = *(const unsigned int*)(qrow + f * 32 + a_koff(j, hi));
    }

    FragC ctx[2][4];
    float mrun[2][8], lrun[2][8];
#pragma unroll
    for (int qt = 0; qt < 2; ++qt) {
#pragma unroll
        for (int dt = 0; dt < 4; ++dt)
#pragma unroll
            for (int i = 0; i < 8; ++i) ctx[qt][dt].f[i] = 0.f;
#pragma unroll
        for (int i = 0; i < 8; ++i) { mrun[qt][i] = -3.0e38f; lrun[qt][i] = 0.f; }
    }

    const int vsr = tid >> 2;            // 0..31 key row for V staging
    const int vsc = (tid & 3) * 16;      // 0,16,32,48 column chunk
#if USE_TDM
    const unsigned int ks_off = (unsigned int)(size_t)&Ks[0][0];  // LDS byte offset
#endif

    for (int kb = 0; kb < 2048; kb += 32) {
        // ---- stage K via Tensor Data Mover; V transposed by VALU ----
#if USE_TDM
        if (wv == 0) tdm_load_ktile(Kh + (size_t)kb * 64, ks_off);
#else
        {
            const bf16_t* kg = Kh + (size_t)(kb + vsr) * 64 + vsc;
            uint4 k0 = ((const uint4*)kg)[0];
            uint4 k1 = ((const uint4*)kg)[1];
            unsigned int* p = (unsigned int*)&Ks[vsr][vsc];
            p[0] = k0.x; p[1] = k0.y; p[2] = k0.z; p[3] = k0.w;
            p[4] = k1.x; p[5] = k1.y; p[6] = k1.z; p[7] = k1.w;
        }
#endif
        {
            const bf16_t* vg = Vh + (size_t)(kb + vsr) * 64 + vsc;
            union { uint4 u[2]; bf16_t e[16]; } vt;
            vt.u[0] = ((const uint4*)vg)[0];
            vt.u[1] = ((const uint4*)vg)[1];
#pragma unroll
            for (int j = 0; j < 16; ++j) Vt[vsc + j][vsr] = vt.e[j];
        }
#if USE_TDM
        if (wv == 0) __builtin_amdgcn_s_wait_tensorcnt(0);
#endif
        __syncthreads();

        if (kb + 32 < 2048) {   // L2 -> WGP prefetch of next key block
            __builtin_prefetch(Kh + (size_t)(kb + 32) * 64, 0, 1);
            __builtin_prefetch(Vh + (size_t)(kb + 32) * 64, 0, 1);
        }

        // ---- scores: S = Q (32x64) x K^T (64x32) -> 2x2 C tiles ----
        FragC sco[2][2];
#pragma unroll
        for (int qt = 0; qt < 2; ++qt)
#pragma unroll
            for (int nt = 0; nt < 2; ++nt)
#pragma unroll
                for (int i = 0; i < 8; ++i) sco[qt][nt].f[i] = 0.f;
#pragma unroll
        for (int nt = 0; nt < 2; ++nt)
#pragma unroll
            for (int f = 0; f < 2; ++f) {
                FragAB bf;
#pragma unroll
                for (int j = 0; j < 8; ++j)
                    bf.u[j] = *(const unsigned int*)&Ks[nt * 16 + lh][f * 32 + hi * 16 + 2 * j];
                sco[0][nt].v = wmma_bf16(qa[0][f], bf, sco[0][nt].v);
                sco[1][nt].v = wmma_bf16(qa[1][f], bf, sco[1][nt].v);
            }

        // ---- online softmax (row stats in registers; width-16 shuffles) ----
        const float bias0 = cbias[kb + lh];
        const float bias1 = cbias[kb + 16 + lh];
#pragma unroll
        for (int qt = 0; qt < 2; ++qt)
#pragma unroll
        for (int i = 0; i < 8; ++i) {
            float v0 = sco[qt][0].f[i] * 0.125f + bias0;
            float v1 = sco[qt][1].f[i] * 0.125f + bias1;
            float mx = fmaxf(v0, v1);
            mx = fmaxf(mx, __shfl_xor(mx, 8, 16));
            mx = fmaxf(mx, __shfl_xor(mx, 4, 16));
            mx = fmaxf(mx, __shfl_xor(mx, 2, 16));
            mx = fmaxf(mx, __shfl_xor(mx, 1, 16));
            const float mn    = fmaxf(mrun[qt][i], mx);
            const float alpha = __expf(mrun[qt][i] - mn);
            const float p0 = __expf(v0 - mn);
            const float p1 = __expf(v1 - mn);
            float rs = p0 + p1;
            rs += __shfl_xor(rs, 8, 16);
            rs += __shfl_xor(rs, 4, 16);
            rs += __shfl_xor(rs, 2, 16);
            rs += __shfl_xor(rs, 1, 16);
            lrun[qt][i] = lrun[qt][i] * alpha + rs;
            mrun[qt][i] = mn;
            ctx[qt][0].f[i] *= alpha; ctx[qt][1].f[i] *= alpha;
            ctx[qt][2].f[i] *= alpha; ctx[qt][3].f[i] *= alpha;
            const int r = qt * 16 + hi * 8 + i;
            Ps[wv][r][lh]      = (bf16_t)p0;
            Ps[wv][r][16 + lh] = (bf16_t)p1;
        }

        // wave-local LDS RAW: DS ops are in-order per wave; fence the compiler
        __builtin_amdgcn_wave_barrier();
        asm volatile("s_wait_dscnt 0x0" ::: "memory");

        // ---- ctx += P (32x32) x V (32x64) ----
        FragAB pa[2];
#pragma unroll
        for (int qt = 0; qt < 2; ++qt)
#pragma unroll
            for (int j = 0; j < 8; ++j)
                pa[qt].u[j] = *(const unsigned int*)&Ps[wv][qt * 16 + lh][a_koff(j, hi)];
#pragma unroll
        for (int dt = 0; dt < 4; ++dt) {
            FragAB vbf;
#pragma unroll
            for (int j = 0; j < 8; ++j)
                vbf.u[j] = *(const unsigned int*)&Vt[dt * 16 + lh][hi * 16 + 2 * j];
            ctx[0][dt].v = wmma_bf16(pa[0], vbf, ctx[0][dt].v);
            ctx[1][dt].v = wmma_bf16(pa[1], vbf, ctx[1][dt].v);
        }
        __syncthreads();
    }

    // ---- epilogue: normalize and store fp32 [B,S,H] ----
#pragma unroll
    for (int qt = 0; qt < 2; ++qt)
#pragma unroll
    for (int dt = 0; dt < 4; ++dt)
#pragma unroll
    for (int i = 0; i < 8; ++i) {
        const int s = q0 + wv * 32 + qt * 16 + hi * 8 + i;
        out[((size_t)b * 2048 + s) * 1024 + h * 64 + dt * 16 + lh] =
            ctx[qt][dt].f[i] / lrun[qt][i];
    }
}

// ---------------------------------------------------------------------------
// launch
// ---------------------------------------------------------------------------
extern "C" void kernel_launch(void* const* d_in, const int* in_sizes, int n_in,
                              void* d_out, int out_size, void* d_ws, size_t ws_size,
                              hipStream_t stream) {
    const float* X     = (const float*)d_in[0];
    const float* cbias = (const float*)d_in[1];
    const float* Wq    = (const float*)d_in[2];
    const float* bq    = (const float*)d_in[3];
    const float* Wk    = (const float*)d_in[4];
    const float* bk    = (const float*)d_in[5];
    const float* Wv    = (const float*)d_in[6];
    const float* bv    = (const float*)d_in[7];
    float* out = (float*)d_out;

    const size_t BS = 4096, H = 1024;
    bf16_t* Xb  = (bf16_t*)d_ws;                 // [4096][1024]
    bf16_t* Wqb = Xb  + BS * H;
    bf16_t* Wkb = Wqb + H * H;
    bf16_t* Wvb = Wkb + H * H;
    bf16_t* Qb  = Wvb + H * H;                   // head-major [b][h][s][hd]
    bf16_t* Kb  = Qb  + BS * H;
    bf16_t* Vb  = Kb  + BS * H;

    cvt_f32_bf16<<<2048, 256, 0, stream>>>(X,  Xb,  (int)(BS * H));
    cvt_f32_bf16<<<1024, 256, 0, stream>>>(Wq, Wqb, (int)(H * H));
    cvt_f32_bf16<<<1024, 256, 0, stream>>>(Wk, Wkb, (int)(H * H));
    cvt_f32_bf16<<<1024, 256, 0, stream>>>(Wv, Wvb, (int)(H * H));

    qkv_gemm<<<dim3(32, 8, 3), 256, 0, stream>>>(Xb, Wqb, Wkb, Wvb,
                                                 bq, bk, bv, Qb, Kb, Vb);

    attn<<<dim3(16, 32), 128, 0, stream>>>(Qb, Kb, Vb, cbias, out);
}
